// QuantumAttentionMechanism_21938692947976
// MI455X (gfx1250) — matrix operand order repaired and checked by reading, not compiled, not run on hardware
//
#include <hip/hip_runtime.h>
#include <hip/hip_bf16.h>

// ---------------------------------------------------------------------------
// QuantumAttentionMechanism for MI455X (gfx1250, wave32, WMMA).
//
// Pipeline (all GEMMs via v_wmma_f32_16x16x32_f16, f32 accumulate):
//   1. fuse_weights:  Wq' = blockdiag(Wqq) @ Wq  (f16), same for K,V
//   2. qq = query @ Wq'^T + bqq   (NT GEMM, f32 in -> f16 out)     x3
//   3. bias = cos_m(W) per head:  A = W@W ; M2 = c3*A@A + c2*A + c1*I ;
//      P = A@M2 + I  -> written f32 straight into d_out w[b=0] slice
//   4. scores: w[b] = qscale*(qq_b @ kq_b^T) + P   (b = 3,2,1,0; b=0 reads
//      its bias element then overwrites it in the same thread -> race-free)
//   5. row softmax in place on w
//   6. attended = w @ vq   (NN, f32 A side converted in loader) -> f16
//   7. out = attended @ Wout^T + bout -> f32 d_out[0:4M]
// ---------------------------------------------------------------------------

typedef __attribute__((ext_vector_type(16))) _Float16 v16h;
typedef __attribute__((ext_vector_type(8)))  float    v8f;

__device__ __forceinline__ v8f wmma32(v16h a, v16h b, v8f c) {
  // 8 args: (neg_a, A, neg_b, B, c_mod, C, reuse_a, reuse_b)
  return __builtin_amdgcn_wmma_f32_16x16x32_f16(
      false, a, false, b, (short)0, c, false, false);
}

// ---- fragment loaders (contiguous k order, same for A and B) --------------------------
// A 16x32 (MxK): lane m = lane&15; lanes 0-15 hold K 0-15,
//                lanes 16-31 hold K 16-31.
template <typename T>
__device__ __forceinline__ v16h load_frag_A(const T* p, int ld, int m0, int k0,
                                            int lane) {
  int m  = m0 + (lane & 15);
  int kb = k0 + ((lane >> 4) << 4);
  const T* q = p + (long long)m * ld + kb;
  v16h f;
#pragma unroll
  for (int i = 0; i < 8; ++i) {
    f[i]     = (_Float16)q[i];        // K = kb .. kb+7
    f[8 + i] = (_Float16)q[8 + i];    // K = kb+8 .. kb+15
  }
  return f;
}

// B 32x16 (KxN): lane n = lane&15; element i -> K = (lane>>4)*16 + i.
// NT: B source is row-major (N,K)  => contiguous 16 halves per lane.
template <typename T>
__device__ __forceinline__ v16h load_frag_B_nt(const T* p, int ld, int n0,
                                               int k0, int lane) {
  int n  = n0 + (lane & 15);
  int kb = k0 + ((lane >> 4) << 4);
  const T* q = p + (long long)n * ld + kb;
  v16h f;
#pragma unroll
  for (int i = 0; i < 16; ++i) f[i] = (_Float16)q[i];
  return f;
}

// NN: B source is row-major (K,N) => per-k loads, coalesced across lanes.
template <typename T>
__device__ __forceinline__ v16h load_frag_B_nn(const T* p, int ld, int n0,
                                               int k0, int lane) {
  int n  = n0 + (lane & 15);
  int kb = k0 + ((lane >> 4) << 4);
  const T* q = p + (long long)kb * ld + n;
  v16h f;
#pragma unroll
  for (int i = 0; i < 16; ++i) f[i] = (_Float16)q[(long long)i * ld];
  return f;
}

// C/D 16x16 f32: VGPR v, lane l -> (M = v + 8*(l>>4), N = l&15)
template <bool HAS_ADD, bool HAS_BIAS, bool HAS_DIAG,
          typename TADD, typename TOUT>
__device__ __forceinline__ void store_tile(v8f c, int m0, int n0, int lane,
                                           TOUT* C, int ldc, const TADD* Add,
                                           int ldadd, float addscale,
                                           const float* biasVec, int biasMod,
                                           float alphaEff, float diagVal) {
  int col   = n0 + (lane & 15);
  int rbase = m0 + ((lane >> 4) << 3);
  float bv = 0.f;
  if constexpr (HAS_BIAS) bv = biasVec[col & (biasMod - 1)];
#pragma unroll
  for (int v = 0; v < 8; ++v) {
    int row   = rbase + v;
    float val = alphaEff * c[v];
    if constexpr (HAS_ADD)
      val += addscale * (float)Add[(long long)row * ldadd + col];
    if constexpr (HAS_BIAS) val += bv;
    if constexpr (HAS_DIAG) {
      if (row == col) val += diagVal;
    }
    C[(long long)row * ldc + col] = (TOUT)val;
  }
}

// ---- generic batched WMMA GEMM: C = alphaEff*(A op(B)) [+ addscale*Add]
//                                   [+ bias[col & (mod-1)]] [+ diagVal*I] ---
// Block: 128 threads = 4 waves in 2x2; wave tile 32x32; block tile 64x64.
// M,N multiples of 64 (grid covers them), K multiple of 64. Assumed exact.
template <typename TA, typename TB, int B_NT, typename TADD, typename TOUT,
          bool HAS_ADD, bool HAS_BIAS, bool HAS_DIAG, bool HAS_APTR>
__global__ void __launch_bounds__(128)
gemm_wmma(const TA* Abase, long long sA, int lda,
          const TB* Bbase, long long sB, int ldb,
          TOUT* Cbase, long long sC, int ldc,
          const TADD* AddBase, long long sAdd, int ldadd, float addscale,
          const float* biasVec, int biasMod,
          float alpha, const float* alphaPtr, float diagVal, int K) {
  const int lane = threadIdx.x & 31;
  const int wave = threadIdx.x >> 5;
  const int m0 = blockIdx.y * 64 + (wave >> 1) * 32;
  const int n0 = blockIdx.x * 64 + (wave & 1) * 32;

  const TA* A = Abase + sA * blockIdx.z;
  const TB* B = Bbase + sB * blockIdx.z;
  TOUT*     C = Cbase + sC * blockIdx.z;
  const TADD* Add = HAS_ADD ? (AddBase + sAdd * blockIdx.z) : nullptr;

  v8f c00 = {}, c01 = {}, c10 = {}, c11 = {};

  for (int k0 = 0; k0 < K; k0 += 64) {
    // prefetch next K block of the streaming operands (global_prefetch_b8)
    if (k0 + 64 < K) {
      __builtin_prefetch(A + (long long)(m0 + (lane & 15)) * lda + k0 + 64,
                         0, 3);
      if constexpr (B_NT)
        __builtin_prefetch(B + (long long)(n0 + (lane & 15)) * ldb + k0 + 64,
                           0, 3);
      else
        __builtin_prefetch(B + (long long)(k0 + 64 + (lane >> 4)) * ldb + n0 +
                               (lane & 15),
                           0, 3);
    }
#pragma unroll
    for (int ks = 0; ks < 2; ++ks) {
      const int kk = k0 + ks * 32;
      v16h a0 = load_frag_A(A, lda, m0,      kk, lane);
      v16h a1 = load_frag_A(A, lda, m0 + 16, kk, lane);
      v16h b0, b1;
      if constexpr (B_NT) {
        b0 = load_frag_B_nt(B, ldb, n0,      kk, lane);
        b1 = load_frag_B_nt(B, ldb, n0 + 16, kk, lane);
      } else {
        b0 = load_frag_B_nn(B, ldb, n0,      kk, lane);
        b1 = load_frag_B_nn(B, ldb, n0 + 16, kk, lane);
      }
      c00 = wmma32(a0, b0, c00);
      c01 = wmma32(a0, b1, c01);
      c10 = wmma32(a1, b0, c10);
      c11 = wmma32(a1, b1, c11);
    }
  }

  float alphaEff = alpha;
  if constexpr (HAS_APTR) alphaEff *= alphaPtr[0];

  store_tile<HAS_ADD, HAS_BIAS, HAS_DIAG>(c00, m0, n0, lane, C, ldc, Add,
                                          ldadd, addscale, biasVec, biasMod,
                                          alphaEff, diagVal);
  store_tile<HAS_ADD, HAS_BIAS, HAS_DIAG>(c01, m0, n0 + 16, lane, C, ldc, Add,
                                          ldadd, addscale, biasVec, biasMod,
                                          alphaEff, diagVal);
  store_tile<HAS_ADD, HAS_BIAS, HAS_DIAG>(c10, m0 + 16, n0, lane, C, ldc, Add,
                                          ldadd, addscale, biasVec, biasMod,
                                          alphaEff, diagVal);
  store_tile<HAS_ADD, HAS_BIAS, HAS_DIAG>(c11, m0 + 16, n0 + 16, lane, C, ldc,
                                          Add, ldadd, addscale, biasVec,
                                          biasMod, alphaEff, diagVal);
}

// ---- Wfused[h*64+i, c] = sum_j Wsmall[i,j] * Wbig[h*64+j, c]  (E=1024) ----
__global__ void __launch_bounds__(256)
fuse_weights(const float* Wbig, const float* Wsmall, _Float16* out) {
  int c = blockIdx.x * 256 + threadIdx.x;
  int r = blockIdx.y;
  int h = r >> 6, i = r & 63;
  float acc = 0.f;
#pragma unroll 8
  for (int j = 0; j < 64; ++j)
    acc += Wsmall[i * 64 + j] * Wbig[(h * 64 + j) * 1024 + c];
  out[(long long)r * 1024 + c] = (_Float16)acc;
}

// ---- in-place row softmax -------------------------------------------------
__global__ void __launch_bounds__(256)
softmax_rows(float* w, int ncols) {
  __shared__ float red[256];
  int tid = threadIdx.x;
  float* row = w + (size_t)blockIdx.x * (size_t)ncols;

  float mx = -3.402823e38f;
  for (int c = tid; c < ncols; c += 256) mx = fmaxf(mx, row[c]);
  red[tid] = mx;
  __syncthreads();
  for (int s = 128; s > 0; s >>= 1) {
    if (tid < s) red[tid] = fmaxf(red[tid], red[tid + s]);
    __syncthreads();
  }
  float m = red[0];
  __syncthreads();

  float sum = 0.f;
  for (int c = tid; c < ncols; c += 256) {
    float e = __expf(row[c] - m);
    row[c]  = e;
    sum += e;
  }
  red[tid] = sum;
  __syncthreads();
  for (int s = 128; s > 0; s >>= 1) {
    if (tid < s) red[tid] += red[tid + s];
    __syncthreads();
  }
  float inv = 1.f / red[0];
  for (int c = tid; c < ncols; c += 256) row[c] *= inv;
}

// ---------------------------------------------------------------------------
extern "C" void kernel_launch(void* const* d_in, const int* in_sizes, int n_in,
                              void* d_out, int out_size, void* d_ws,
                              size_t ws_size, hipStream_t stream) {
  (void)in_sizes; (void)n_in; (void)out_size; (void)ws_size;

  const float* query  = (const float*)d_in[0];
  const float* key_t  = (const float*)d_in[1];
  const float* value  = (const float*)d_in[2];
  const float* Wq     = (const float*)d_in[3];
  const float* Wk     = (const float*)d_in[4];
  const float* Wv     = (const float*)d_in[5];
  const float* Wqq    = (const float*)d_in[6];
  const float* bqq    = (const float*)d_in[7];
  const float* Wkq    = (const float*)d_in[8];
  const float* bkq    = (const float*)d_in[9];
  const float* Wvq    = (const float*)d_in[10];
  const float* bvq    = (const float*)d_in[11];
  const float* interf = (const float*)d_in[12];
  const float* qscale = (const float*)d_in[13];
  const float* Wout   = (const float*)d_in[14];
  const float* bout   = (const float*)d_in[15];

  const long long MM   = 1024LL * 1024LL;  // 1M elements
  float* outp = (float*)d_out;             // (4,1024,1024)
  float* w    = outp + 4 * MM;             // (4,16,1024,1024)

  char* ws = (char*)d_ws;
  const size_t MB = 1u << 20;
  _Float16* fq    = (_Float16*)(ws + 0 * MB);    // 2 MB fused Wq'
  _Float16* fk    = (_Float16*)(ws + 2 * MB);
  _Float16* fv    = (_Float16*)(ws + 4 * MB);
  _Float16* qq    = (_Float16*)(ws + 6 * MB);    // (4096,1024) f16, 8 MB
  _Float16* kq    = (_Float16*)(ws + 14 * MB);
  _Float16* vq    = (_Float16*)(ws + 22 * MB);
  _Float16* attnd = (_Float16*)(ws + 30 * MB);   // (4096,1024) f16
  _Float16* Abuf  = (_Float16*)(ws + 38 * MB);   // (16,1024,1024) f16, 32 MB
  _Float16* M2buf = (_Float16*)(ws + 70 * MB);   // (16,1024,1024) f16, 32 MB

  dim3 blk(128);

  // 1. fuse quantum projections into the QKV weights
  fuse_weights<<<dim3(4, 1024), 256, 0, stream>>>(Wq, Wqq, fq);
  fuse_weights<<<dim3(4, 1024), 256, 0, stream>>>(Wk, Wkq, fk);
  fuse_weights<<<dim3(4, 1024), 256, 0, stream>>>(Wv, Wvq, fv);

  // 2. qq/kq/vq = X @ Wf^T + b[col%64]   (M=4096,N=1024,K=1024)
  gemm_wmma<float, _Float16, 1, float, _Float16, false, true, false, false>
      <<<dim3(16, 64, 1), blk, 0, stream>>>(
          query, 0, 1024, fq, 0, 1024, qq, 0, 1024,
          (const float*)nullptr, 0, 1, 0.f, bqq, 64, 1.f, nullptr, 0.f, 1024);
  gemm_wmma<float, _Float16, 1, float, _Float16, false, true, false, false>
      <<<dim3(16, 64, 1), blk, 0, stream>>>(
          key_t, 0, 1024, fk, 0, 1024, kq, 0, 1024,
          (const float*)nullptr, 0, 1, 0.f, bkq, 64, 1.f, nullptr, 0.f, 1024);
  gemm_wmma<float, _Float16, 1, float, _Float16, false, true, false, false>
      <<<dim3(16, 64, 1), blk, 0, stream>>>(
          value, 0, 1024, fv, 0, 1024, vq, 0, 1024,
          (const float*)nullptr, 0, 1, 0.f, bvq, 64, 1.f, nullptr, 0.f, 1024);

  // 3. matrix cosine per head: bias = I - A/2 + A^2/24 - A^3/720, A = W^2
  const float c1 = -0.5f, c2 = 1.f / 24.f, c3 = -1.f / 720.f;
  // A = W @ W (NN, f32 sources)
  gemm_wmma<float, float, 0, float, _Float16, false, false, false, false>
      <<<dim3(16, 16, 16), blk, 0, stream>>>(
          interf, MM, 1024, interf, MM, 1024, Abuf, MM, 1024,
          (const float*)nullptr, 0, 1, 0.f, nullptr, 1, 1.f, nullptr, 0.f,
          1024);
  // M2 = c3*(A@A) + c2*A + c1*I
  gemm_wmma<_Float16, _Float16, 0, _Float16, _Float16, true, false, true,
            false>
      <<<dim3(16, 16, 16), blk, 0, stream>>>(
          Abuf, MM, 1024, Abuf, MM, 1024, M2buf, MM, 1024,
          Abuf, MM, 1024, c2, nullptr, 1, c3, nullptr, c1, 1024);
  // P = A@M2 + I  -> f32, stored into w[b=0] slice of d_out
  gemm_wmma<_Float16, _Float16, 0, float, float, false, false, true, false>
      <<<dim3(16, 16, 16), blk, 0, stream>>>(
          Abuf, MM, 1024, M2buf, MM, 1024, w, MM, 1024,
          (const float*)nullptr, 0, 1, 0.f, nullptr, 1, 1.f, nullptr, 1.0f,
          1024);

  // 4. scores: w[b,h] = qscale*(qq_bh @ kq_bh^T) + P[h]; b=0 last (in-place)
  const int border[4] = {3, 2, 1, 0};
  for (int bi = 0; bi < 4; ++bi) {
    int b = border[bi];
    gemm_wmma<_Float16, _Float16, 1, float, float, true, false, false, true>
        <<<dim3(16, 16, 16), blk, 0, stream>>>(
            qq + (long long)b * MM, 64, 1024,   // head h -> column offset 64h
            kq + (long long)b * MM, 64, 1024,
            w + (long long)b * 16 * MM, MM, 1024,
            w, MM, 1024, 1.0f,                  // bias P[h] from w[b=0] slice
            nullptr, 1, 1.0f, qscale, 0.f, 64);
  }

  // 5. softmax over last dim, in place (4*16*1024 rows)
  softmax_rows<<<65536, 256, 0, stream>>>(w, 1024);

  // 6. attended[b,h] = w[b,h] @ vq[b,h]   (M=1024,N=64,K=1024, NN)
  for (int b = 0; b < 4; ++b) {
    gemm_wmma<float, _Float16, 0, float, _Float16, false, false, false, false>
        <<<dim3(1, 16, 16), blk, 0, stream>>>(
            w + (long long)b * 16 * MM, MM, 1024,
            vq + (long long)b * MM, 64, 1024,
            attnd + (long long)b * MM, 64, 1024,
            (const float*)nullptr, 0, 1, 0.f, nullptr, 1, 1.f, nullptr, 0.f,
            1024);
  }

  // 7. out = attended @ Wout^T + bout  (f32 result)
  gemm_wmma<_Float16, float, 1, float, float, false, true, false, false>
      <<<dim3(16, 64, 1), blk, 0, stream>>>(
          attnd, 0, 1024, Wout, 0, 1024, outp, 0, 1024,
          (const float*)nullptr, 0, 1, 0.f, bout, 1024, 1.f, nullptr, 0.f,
          1024);
}
